// GraphSage_6605659701635
// MI455X (gfx1250) — matrix operand order repaired and checked by reading, not compiled, not run on hardware
//
#include <hip/hip_runtime.h>
#include <hip/hip_bf16.h>
#include <stdint.h>

// ---------------- WMMA types ----------------
typedef __attribute__((ext_vector_type(16))) __bf16 v16bf;
typedef __attribute__((ext_vector_type(8)))  float  v8f;

#define BM 128
#define BN 64
#define BK 32
#define LDSK 40  // padded LDS row stride in ushorts (80B: 16B-aligned, conflict-free)

__device__ __forceinline__ unsigned short f2bf(float f) {
  uint32_t u = __float_as_uint(f);
  u += 0x7FFFu + ((u >> 16) & 1u);   // round-to-nearest-even
  return (unsigned short)(u >> 16);
}

__device__ __forceinline__ float elu1(float v) {
  return v > 0.f ? v : (__expf(v) - 1.f);
}

union FragU {
  uint4 q[2];
  v16bf v;
};

// A-matrix 16x32 bf16 fragment: lane m = lane&15, khalf = lane>>4
// element e -> K = (e>>3)*16 + khalf*8 + (e&7)
__device__ __forceinline__ v16bf load_fragA(const unsigned short* base, int row0, int lane) {
  const int m = lane & 15, kh = lane >> 4;
  const unsigned short* p = base + (row0 + m) * LDSK + kh * 8;
  FragU f;
  f.q[0] = *(const uint4*)p;          // K = kh*8 + 0..7
  f.q[1] = *(const uint4*)(p + 16);   // K = 16 + kh*8 + 0..7
  return f.v;
}

// B-matrix 32x16 bf16 fragment (LDS holds B transposed [N][K]):
// lane n = lane&15, khalf = lane>>4 ; element e -> K = (e>>3)*16 + khalf*8 + (e&7)
__device__ __forceinline__ v16bf load_fragB(const unsigned short* base, int col0, int lane) {
  const int n = lane & 15, kh = lane >> 4;
  const unsigned short* p = base + (col0 + n) * LDSK + kh * 8;
  FragU f;
  f.q[0] = *(const uint4*)p;          // K = kh*8 + 0..7
  f.q[1] = *(const uint4*)(p + 16);   // K = 16 + kh*8 + 0..7
  return f.v;
}

// One BK=32 step of the 2x2 WMMA tile for this wave.
__device__ __forceinline__ void mma_step(const unsigned short* As, const unsigned short* Bs,
                                         int wr, int wc, int lane, v8f (&acc)[2][2]) {
  v16bf af[2], bfr[2];
#pragma unroll
  for (int mt = 0; mt < 2; ++mt) af[mt]  = load_fragA(As, wr * 32 + mt * 16, lane);
#pragma unroll
  for (int nt = 0; nt < 2; ++nt) bfr[nt] = load_fragB(Bs, wc * 32 + nt * 16, lane);
#pragma unroll
  for (int mt = 0; mt < 2; ++mt)
#pragma unroll
    for (int nt = 0; nt < 2; ++nt)
      acc[mt][nt] = __builtin_amdgcn_wmma_f32_16x16x32_bf16(
          false, af[mt], false, bfr[nt], (short)0, acc[mt][nt], false, false);
}

// ---------------- GEMM ----------------
// C[M, Ntot] = A[M,K] @ [B0 | B1]  (B0,B1 each [K,N0]; B1 may be null, then Ntot==N0)
// Optional bias[Ntot] and ELU. Requires N0 % BN == 0 when B1 != null.
__global__ __launch_bounds__(256) void gemm_wmma_bf16(
    const float* __restrict__ A, const float* __restrict__ B0,
    const float* __restrict__ B1, float* __restrict__ C,
    const float* __restrict__ bias,
    int M, int K, int N0, int Ntot, int act)
{
  __shared__ unsigned short As[BM * LDSK];   // [BM][BK] bf16
  __shared__ unsigned short Bs[BN * LDSK];   // [BN][BK] bf16 (transposed)

  const int tid  = threadIdx.x;
  const int lane = tid & 31;
  const int wave = tid >> 5;           // 8 waves
  const int wr   = wave >> 1;          // 0..3 : 32-row strip
  const int wc   = wave & 1;           // 0..1 : 32-col strip
  const int blockRow = blockIdx.y * BM;
  const int blockCol = blockIdx.x * BN;

  // Select which weight matrix this 64-col block reads (never straddles the seam).
  const float* Bsel = B0;
  int colOff = blockCol;
  if (B1 && blockCol >= N0) { Bsel = B1; colOff = blockCol - N0; }

  const bool rowsFull = (blockRow + BM <= M);
  const bool colsFull = (colOff + BN <= N0) && (blockCol + BN <= Ntot);

  const v8f zerov = {0.f, 0.f, 0.f, 0.f, 0.f, 0.f, 0.f, 0.f};
  v8f acc[2][2];
#pragma unroll
  for (int i = 0; i < 2; ++i)
#pragma unroll
    for (int j = 0; j < 2; ++j) acc[i][j] = zerov;

  int k0 = 0;
  const int kMain = K & ~(BK - 1);

  // ---- Main loop: interior blocks, full K-tiles, wide loads only ----
  if (rowsFull && colsFull) {
    for (; k0 < kMain; k0 += BK) {
      // A tile: 16B global loads, pack 4 bf16 -> one b64 LDS store
#pragma unroll
      for (int i = 0; i < 4; ++i) {
        const int linear = i * 256 + tid;        // 0..1023 float4 slots
        const int r = linear >> 3, f4 = linear & 7;
        float4 v;
        __builtin_memcpy(&v, A + (long)(blockRow + r) * K + k0 + f4 * 4, 16);
        uint2 pk;
        pk.x = ((unsigned)f2bf(v.y) << 16) | f2bf(v.x);
        pk.y = ((unsigned)f2bf(v.w) << 16) | f2bf(v.z);
        *(uint2*)(As + r * LDSK + f4 * 4) = pk;
      }
      if (k0 + BK < K)  // prefetch next K-tile of A (global_prefetch_b8)
        __builtin_prefetch(A + (long)(blockRow + (tid >> 1)) * K + (k0 + BK) + (tid & 1) * 16, 0, 3);

      // B tile: 16B global loads along N, transpose into [n][k]
#pragma unroll
      for (int i = 0; i < 2; ++i) {
        const int linear = i * 256 + tid;        // 0..511 float4 slots
        const int k = linear >> 4, nf4 = linear & 15;
        float4 v;
        __builtin_memcpy(&v, Bsel + (long)(k0 + k) * N0 + colOff + nf4 * 4, 16);
        const int nb = nf4 * 4;
        Bs[(nb + 0) * LDSK + k] = f2bf(v.x);
        Bs[(nb + 1) * LDSK + k] = f2bf(v.y);
        Bs[(nb + 2) * LDSK + k] = f2bf(v.z);
        Bs[(nb + 3) * LDSK + k] = f2bf(v.w);
      }
      __syncthreads();
      mma_step(As, Bs, wr, wc, lane, acc);
      __syncthreads();
    }
  }

  // ---- Generic loop: K tail and edge blocks (branch-free clamped loads) ----
  for (; k0 < K; k0 += BK) {
#pragma unroll
    for (int i = 0; i < 16; ++i) {
      const int linear = i * 256 + tid;          // 0..4095
      const int r = linear >> 5, k = linear & 31;
      const int gr = blockRow + r, gk = k0 + k;
      const int grc = gr < M ? gr : M - 1;
      const int gkc = gk < K ? gk : K - 1;
      float v = A[(long)grc * K + gkc];
      v = (gr < M && gk < K) ? v : 0.f;
      As[r * LDSK + k] = f2bf(v);
    }
#pragma unroll
    for (int i = 0; i < 8; ++i) {
      const int linear = i * 256 + tid;          // 0..2047
      const int n = linear >> 5, k = linear & 31;
      const int gk = k0 + k;
      const int gc = blockCol + n;
      const float* bp = B0;
      int cc = gc;
      if (B1 && gc >= N0) { bp = B1; cc = gc - N0; }
      const int gkc = gk < K ? gk : K - 1;
      const int ccc = cc < N0 ? cc : N0 - 1;
      float v = bp[(long)gkc * N0 + ccc];
      v = (gk < K && gc < Ntot && cc < N0) ? v : 0.f;
      Bs[n * LDSK + k] = f2bf(v);
    }
    __syncthreads();
    mma_step(As, Bs, wr, wc, lane, acc);
    __syncthreads();
  }

  // Epilogue: C/D layout -> VGPR r: M = r + 8*(lane>>4), N = lane&15
#pragma unroll
  for (int mt = 0; mt < 2; ++mt) {
#pragma unroll
    for (int nt = 0; nt < 2; ++nt) {
      const int col   = blockCol + wc * 32 + nt * 16 + (lane & 15);
      const int rbase = blockRow + wr * 32 + mt * 16 + ((lane >> 4) << 3);
      if (col < Ntot) {
        const float bv = bias ? bias[col] : 0.f;
#pragma unroll
        for (int r = 0; r < 8; ++r) {
          const int row = rbase + r;
          if (row < M) {
            float v = acc[mt][nt][r] + bv;
            if (act) v = elu1(v);
            C[(long)row * Ntot + col] = v;
          }
        }
      }
    }
  }
}

// ---------------- Graph kernels ----------------
__global__ void zero_kernel(float* __restrict__ p, long n) {
  long i = (long)blockIdx.x * blockDim.x + threadIdx.x;
  const long stride = (long)gridDim.x * blockDim.x;
  for (; i < n; i += stride) p[i] = 0.f;
}

__global__ void deg_kernel(const int* __restrict__ dst, float* __restrict__ deg, int E) {
  const int e = blockIdx.x * blockDim.x + threadIdx.x;
  if (e < E) atomicAdd(&deg[dst[e]], 1.0f);
}

// msg[dst, 0:256] += h[src, 0:256] where h rows have stride hstride.
__global__ void scatter_kernel(const float* __restrict__ h, const int* __restrict__ src,
                               const int* __restrict__ dst, float* __restrict__ msg,
                               int E, int hstride) {
  const long idx = (long)blockIdx.x * blockDim.x + threadIdx.x;
  const long total = (long)E * 64;
  if (idx >= total) return;
  const int e = (int)(idx >> 6);
  const int c = (int)((idx & 63) << 2);
  const int s = src[e], d = dst[e];
  const float4 v = *(const float4*)&h[(long)s * hstride + c];
  float* out = &msg[(long)d * 256 + c];
  atomicAdd(out + 0, v.x);
  atomicAdd(out + 1, v.y);
  atomicAdd(out + 2, v.z);
  atomicAdd(out + 3, v.w);
}

// per-node: val = msg/max(deg,1) + hr + b ; l2-normalize over 256 feats ; ELU
__global__ __launch_bounds__(256) void post_kernel(
    const float* __restrict__ msg, const float* __restrict__ deg,
    const float* __restrict__ hr, int hrstride, const float* __restrict__ bias,
    float* __restrict__ out)
{
  const int node = blockIdx.x;
  const int f = threadIdx.x;             // 256 feats
  const float d = fmaxf(deg[node], 1.0f);
  const float v = msg[(long)node * 256 + f] / d + hr[(long)node * hrstride + f] + bias[f];

  float sq = v * v;
#pragma unroll
  for (int o = 16; o > 0; o >>= 1) sq += __shfl_xor(sq, o, 32);

  __shared__ float red[8];
  const int wave = f >> 5, lane = f & 31;
  if (lane == 0) red[wave] = sq;
  __syncthreads();
  float tot = 0.f;
#pragma unroll
  for (int i = 0; i < 8; ++i) tot += red[i];
  const float norm = fmaxf(sqrtf(tot), 1e-12f);
  out[(long)node * 256 + f] = elu1(v / norm);
}

// ---------------- Host orchestration ----------------
extern "C" void kernel_launch(void* const* d_in, const int* in_sizes, int n_in,
                              void* d_out, int out_size, void* d_ws, size_t ws_size,
                              hipStream_t stream) {
  const float* x    = (const float*)d_in[0];
  const int*   edges= (const int*)d_in[1];
  const float* Wl1  = (const float*)d_in[2];
  const float* Wr1  = (const float*)d_in[3];
  const float* b1   = (const float*)d_in[4];
  const float* Wl2  = (const float*)d_in[5];
  const float* Wr2  = (const float*)d_in[6];
  const float* b2   = (const float*)d_in[7];
  const float* Wl3  = (const float*)d_in[8];
  const float* Wr3  = (const float*)d_in[9];
  const float* b3   = (const float*)d_in[10];
  const float* Wf1  = (const float*)d_in[11];
  const float* bf1  = (const float*)d_in[12];
  const float* Wf2  = (const float*)d_in[13];
  const float* bf2  = (const float*)d_in[14];
  const float* Wf3  = (const float*)d_in[15];
  const float* bf3  = (const float*)d_in[16];
  (void)n_in; (void)out_size; (void)ws_size;

  const int  DH = 256, DO = 100;
  const int  DI = in_sizes[2] / DH;           // 2613
  const long NN = (long)in_sizes[0] / DI;     // 50000
  const int  E  = in_sizes[1] / 2;            // 1600000
  const int* srcv = edges;
  const int* dstv = edges + E;

  // workspace carve-out
  char* ws = (char*)d_ws;
  size_t off = 0;
  auto take = [&](size_t bytes) -> char* {
    char* p = ws + off;
    off += (bytes + 255) & ~(size_t)255;
    return p;
  };
  float* hcat = (float*)take((size_t)NN * 2 * DH * sizeof(float));  // [NN, 512] = [hl | hr]
  float* msg  = (float*)take((size_t)NN * DH * sizeof(float));
  float* h    = (float*)take((size_t)NN * DH * sizeof(float));
  float* t1   = (float*)take((size_t)NN * DH * sizeof(float));
  float* deg  = (float*)take((size_t)NN * sizeof(float));

  const dim3 blk(256);
  auto zero = [&](float* p, long n) {
    zero_kernel<<<2048, 256, 0, stream>>>(p, n);
  };

  // degree (same for all layers)
  zero(deg, NN);
  deg_kernel<<<(E + 255) / 256, 256, 0, stream>>>(dstv, deg, E);

  auto sage = [&](const float* in, int K, const float* Wl, const float* Wr, const float* bb) {
    // hcat = in @ [Wl | Wr]  -> one pass over `in`
    dim3 grid((2 * DH + BN - 1) / BN, (unsigned)((NN + BM - 1) / BM));
    gemm_wmma_bf16<<<grid, blk, 0, stream>>>(in, Wl, Wr, hcat, nullptr,
                                             (int)NN, K, DH, 2 * DH, 0);
    zero(msg, NN * (long)DH);
    const long total = (long)E * 64;
    scatter_kernel<<<(unsigned)((total + 255) / 256), 256, 0, stream>>>(
        hcat, srcv, dstv, msg, E, 2 * DH);
    post_kernel<<<(unsigned)NN, 256, 0, stream>>>(msg, deg, hcat + DH, 2 * DH, bb, h);
  };

  sage(x, DI, Wl1, Wr1, b1);     // layer 1: 2613 -> 256
  sage(h,  DH, Wl2, Wr2, b2);    // layer 2
  sage(h,  DH, Wl3, Wr3, b3);    // layer 3

  // FC stack (bias + ELU fused into GEMM epilogue)
  auto gemm1 = [&](const float* A, const float* B, float* C, const float* bias,
                   int K, int N, int act) {
    dim3 grid((N + BN - 1) / BN, (unsigned)((NN + BM - 1) / BM));
    gemm_wmma_bf16<<<grid, blk, 0, stream>>>(A, B, nullptr, C, bias, (int)NN, K, N, N, act);
  };
  gemm1(h,    Wf1, t1,   bf1, DH, DH, 1);
  gemm1(t1,   Wf2, hcat, bf2, DH, DH, 1);              // reuse hcat as t2 [NN,256]
  gemm1(hcat, Wf3, (float*)d_out, bf3, DH, DO, 0);     // final: 256 -> 100
}
